// SelfAttention_4346506904227
// MI455X (gfx1250) — matrix operand order repaired and checked by reading, not compiled, not run on hardware
//
#include <hip/hip_runtime.h>
#include <hip/hip_bf16.h>

typedef __attribute__((ext_vector_type(16))) __bf16 v16bf;
typedef __attribute__((ext_vector_type(8)))  __bf16 v8bf;
typedef __attribute__((ext_vector_type(8)))  float  v8f;

#define DDIM 512
#define SLEN 4096
#define NBATCH 4

// Native f32 -> bf16 conversion: gfx1250 has bf16 VALU support, so the
// fptrunc lowers to v_cvt instructions instead of manual integer rounding.
static __device__ __forceinline__ __bf16 f2bf(float f) { return (__bf16)f; }

static __device__ __forceinline__ v16bf cat16(v8bf lo, v8bf hi) {
  return __builtin_shufflevector(lo, hi, 0, 1, 2, 3, 4, 5, 6, 7,
                                         8, 9, 10, 11, 12, 13, 14, 15);
}

// ---------------------------------------------------------------------------
// Kernel 0: convert Wq/Wk/Wv (fp32, [K][N]) to bf16 transposed [mat][N][K]
// so B-fragment loads in the projection GEMM are contiguous along K.
// ---------------------------------------------------------------------------
__global__ void prep_weights(const float* __restrict__ Wq,
                             const float* __restrict__ Wk,
                             const float* __restrict__ Wv,
                             __bf16* __restrict__ Wt) {
  int idx = blockIdx.x * blockDim.x + threadIdx.x;   // 0 .. 3*512*512-1
  int mat = idx >> 18;
  int rem = idx & 0x3FFFF;
  int k   = rem >> 9;
  int n   = rem & 511;
  const float* W = (mat == 0) ? Wq : (mat == 1) ? Wk : Wv;
  Wt[(size_t)mat * DDIM * DDIM + (size_t)n * DDIM + k] = f2bf(W[(size_t)k * DDIM + n]);
}

// ---------------------------------------------------------------------------
// Kernel 1: QKV projection. One wave computes one 16x16 tile of one of Q/K/V.
// Q,K written row-major bf16 [B*S][512]; V written transposed [B][512][S]
// so the P*V B-fragments in flash attention are contiguous.
// ---------------------------------------------------------------------------
__global__ void qkv_proj(const float*  __restrict__ x,
                         const __bf16* __restrict__ Wt,
                         const float*  __restrict__ bq,
                         const float*  __restrict__ bk,
                         const float*  __restrict__ bv,
                         __bf16* __restrict__ Qb,
                         __bf16* __restrict__ Kb,
                         __bf16* __restrict__ Vt) {
  int tid = threadIdx.x, wave = tid >> 5, lane = tid & 31;
  int khalf = lane >> 4, lan15 = lane & 15;
  long gw = (long)blockIdx.x * 8 + wave;       // 0 .. 98303
  int rowTile = (int)(gw / 96);                // 0 .. 1023  (B*S/16)
  int rem = (int)(gw % 96);
  int mat = rem >> 5;                          // 0=Q 1=K 2=V
  int colTile = rem & 31;                      // 0 .. 31    (512/16)

  int grow = rowTile * 16 + lan15;             // A-fragment row for this lane
  const float* arow = x + (size_t)grow * DDIM;
  int ncol = colTile * 16 + lan15;             // B-fragment column for this lane
  const __bf16* brow = Wt + (size_t)mat * DDIM * DDIM + (size_t)ncol * DDIM;

  v8f acc = {};
#pragma unroll 4
  for (int kb = 0; kb < DDIM; kb += 32) {
    // A fragment layout: k = kb + 16*khalf + t
    const float4* p0 = (const float4*)(arow + kb + 16 * khalf);
    const float4* p1 = (const float4*)(arow + kb + 16 * khalf + 8);
    float4 a0 = p0[0], a1 = p0[1], a2 = p1[0], a3 = p1[1];
    v8bf lo, hi;
    lo[0]=f2bf(a0.x); lo[1]=f2bf(a0.y); lo[2]=f2bf(a0.z); lo[3]=f2bf(a0.w);
    lo[4]=f2bf(a1.x); lo[5]=f2bf(a1.y); lo[6]=f2bf(a1.z); lo[7]=f2bf(a1.w);
    hi[0]=f2bf(a2.x); hi[1]=f2bf(a2.y); hi[2]=f2bf(a2.z); hi[3]=f2bf(a2.w);
    hi[4]=f2bf(a3.x); hi[5]=f2bf(a3.y); hi[6]=f2bf(a3.z); hi[7]=f2bf(a3.w);
    v16bf a = cat16(lo, hi);
    // B fragment: k = kb + 16*khalf + t -> 16 contiguous bf16 (32 bytes)
    v16bf b = *(const v16bf*)(brow + kb + 16 * khalf);
    acc = __builtin_amdgcn_wmma_f32_16x16x32_bf16(false, a, false, b, (short)0, acc,
                                                  false, false);
  }

  const float* bias = (mat == 0) ? bq : (mat == 1) ? bk : bv;
  float bb = bias[ncol];

  if (mat == 2) {  // V -> transposed layout Vt[b][dim][s]
#pragma unroll
    for (int r = 0; r < 8; r++) {
      int m = r + 8 * khalf;
      int gr = rowTile * 16 + m;
      int bat = gr >> 12, s = gr & (SLEN - 1);
      Vt[(size_t)bat * DDIM * SLEN + (size_t)ncol * SLEN + s] = f2bf(acc[r] + bb);
    }
  } else {
    __bf16* dst = (mat == 0) ? Qb : Kb;
#pragma unroll
    for (int r = 0; r < 8; r++) {
      int m = r + 8 * khalf;
      int gr = rowTile * 16 + m;
      dst[(size_t)gr * DDIM + ncol] = f2bf(acc[r] + bb);
    }
  }
}

// ---------------------------------------------------------------------------
// Kernel 2: flash attention. One block = 16 query rows, 8 waves.
// Wave w owns output dims [64w, 64w+64). Streams 32-key tiles:
//   scores: each wave computes partial Q*K^T over its 64-dim slice (4 WMMA),
//           reduced across waves with ds_add_f32 into LDS,
//   online softmax (one thread per q-row),
//   P*V: 4 WMMA per wave against transposed V (contiguous B-fragments).
// ---------------------------------------------------------------------------
__global__ void __launch_bounds__(256)
flash_attn(const __bf16* __restrict__ Qb,
           const __bf16* __restrict__ Kb,
           const __bf16* __restrict__ Vt,
           float* __restrict__ out) {
  __shared__ __attribute__((aligned(32))) __bf16 Qs[16][DDIM];   // 16 KB
  __shared__ __attribute__((aligned(32))) __bf16 Pb[16][32];     // 1 KB
  __shared__ float sc[16][32];                                   // 2 KB
  __shared__ float row_m[16], row_l[16], row_scale[16];

  int tid = threadIdx.x, wave = tid >> 5, lane = tid & 31;
  int khalf = lane >> 4, lan15 = lane & 15;
  int qt = blockIdx.x;
  int bat = qt >> 8;                 // 256 q-tiles per batch
  int q0 = (qt & 255) * 16;

  const __bf16* Qblk = Qb + ((size_t)bat * SLEN + q0) * DDIM;
  const __bf16* Kbat = Kb + (size_t)bat * SLEN * DDIM;
  const __bf16* Vbat = Vt + (size_t)bat * DDIM * SLEN;

  // Stage the 16x512 bf16 Q tile into LDS.
  for (int i = tid; i < 16 * DDIM / 8; i += 256)
    ((v8bf*)Qs)[i] = ((const v8bf*)Qblk)[i];
  if (tid < 16) { row_m[tid] = -1e30f; row_l[tid] = 0.0f; }

  v8f o0 = {}, o1 = {}, o2 = {}, o3 = {};
  const float ssc = 1.4426950408889634f / 22.62741699796952f;  // log2(e)/sqrt(512)

  for (int kt = 0; kt < SLEN / 32; kt++) {
    __syncthreads();
    ((float*)sc)[tid] = 0.0f;
    ((float*)sc)[tid + 256] = 0.0f;
    __syncthreads();

    // ---- partial scores over this wave's 64-dim slice ----
    v8f s0 = {}, s1 = {};
    const __bf16* k0row = Kbat + (size_t)(kt * 32 + lan15) * DDIM;
    const __bf16* k1row = Kbat + (size_t)(kt * 32 + 16 + lan15) * DDIM;
#pragma unroll
    for (int c = 0; c < 2; c++) {
      int kb = wave * 64 + c * 32;
      v8bf c0 = *(const v8bf*)&Qs[lan15][kb + 16 * khalf];
      v8bf c1 = *(const v8bf*)&Qs[lan15][kb + 16 * khalf + 8];
      v16bf aq = cat16(c0, c1);
      v16bf b0 = *(const v16bf*)(k0row + kb + 16 * khalf);
      v16bf b1 = *(const v16bf*)(k1row + kb + 16 * khalf);
      s0 = __builtin_amdgcn_wmma_f32_16x16x32_bf16(false, aq, false, b0, (short)0, s0,
                                                   false, false);
      s1 = __builtin_amdgcn_wmma_f32_16x16x32_bf16(false, aq, false, b1, (short)0, s1,
                                                   false, false);
    }
    if (kt + 1 < SLEN / 32)
      __builtin_prefetch(Kbat + (size_t)((kt + 1) * 32 + lan15) * DDIM, 0, 1);

#pragma unroll
    for (int r = 0; r < 8; r++) {
      int m = r + 8 * khalf;
      atomicAdd(&sc[m][lan15], s0[r]);
      atomicAdd(&sc[m][16 + lan15], s1[r]);
    }
    __syncthreads();

    // ---- online softmax (one thread per query row) ----
    if (tid < 16) {
      int m = tid;
      float tmax = -1e30f;
      for (int j = 0; j < 32; j++) tmax = fmaxf(tmax, sc[m][j] * ssc);
      float om = row_m[m];
      float nm = fmaxf(om, tmax);
      float f = __builtin_exp2f(om - nm);
      float l = row_l[m] * f;
      for (int j = 0; j < 32; j++) {
        float p = __builtin_exp2f(sc[m][j] * ssc - nm);
        l += p;
        Pb[m][j] = f2bf(p);
      }
      row_m[m] = nm; row_l[m] = l; row_scale[m] = f;
    }
    __syncthreads();

    // ---- rescale accumulators, then accumulate P (16x32) x V (32x64 slice) ----
    v8bf p0 = *(const v8bf*)&Pb[lan15][16 * khalf];
    v8bf p1 = *(const v8bf*)&Pb[lan15][16 * khalf + 8];
    v16bf ap = cat16(p0, p1);
#pragma unroll
    for (int r = 0; r < 8; r++) {
      float f = row_scale[r + 8 * khalf];
      o0[r] *= f; o1[r] *= f; o2[r] *= f; o3[r] *= f;
    }
    const __bf16* vbase = Vbat + (size_t)(wave * 64 + lan15) * SLEN + kt * 32 + 16 * khalf;
    v16bf bv0 = *(const v16bf*)(vbase);
    v16bf bv1 = *(const v16bf*)(vbase + 16 * SLEN);
    v16bf bv2 = *(const v16bf*)(vbase + 32 * SLEN);
    v16bf bv3 = *(const v16bf*)(vbase + 48 * SLEN);
    o0 = __builtin_amdgcn_wmma_f32_16x16x32_bf16(false, ap, false, bv0, (short)0, o0,
                                                 false, false);
    o1 = __builtin_amdgcn_wmma_f32_16x16x32_bf16(false, ap, false, bv1, (short)0, o1,
                                                 false, false);
    o2 = __builtin_amdgcn_wmma_f32_16x16x32_bf16(false, ap, false, bv2, (short)0, o2,
                                                 false, false);
    o3 = __builtin_amdgcn_wmma_f32_16x16x32_bf16(false, ap, false, bv3, (short)0, o3,
                                                 false, false);
  }

  // ---- finalize: divide by softmax denominator, write fp32 output ----
  float* obase = out + ((size_t)bat * SLEN + q0) * DDIM + wave * 64 + lan15;
#pragma unroll
  for (int r = 0; r < 8; r++) {
    int m = r + 8 * khalf;
    float inv = 1.0f / row_l[m];
    float* orow = obase + (size_t)m * DDIM;
    orow[0]  = o0[r] * inv;
    orow[16] = o1[r] * inv;
    orow[32] = o2[r] * inv;
    orow[48] = o3[r] * inv;
  }
}

// ---------------------------------------------------------------------------
extern "C" void kernel_launch(void* const* d_in, const int* in_sizes, int n_in,
                              void* d_out, int out_size, void* d_ws, size_t ws_size,
                              hipStream_t stream) {
  const float* x  = (const float*)d_in[0];
  const float* Wq = (const float*)d_in[1];
  const float* bq = (const float*)d_in[2];
  const float* Wk = (const float*)d_in[3];
  const float* bk = (const float*)d_in[4];
  const float* Wv = (const float*)d_in[5];
  const float* bv = (const float*)d_in[6];
  float* out = (float*)d_out;

  char* ws = (char*)d_ws;
  __bf16* Qb = (__bf16*)(ws);                               // 8 MiB  [B*S][512]
  __bf16* Kb = (__bf16*)(ws + (size_t)8  * 1024 * 1024);    // 8 MiB  [B*S][512]
  __bf16* Vt = (__bf16*)(ws + (size_t)16 * 1024 * 1024);    // 8 MiB  [B][512][S]
  __bf16* Wt = (__bf16*)(ws + (size_t)24 * 1024 * 1024);    // 1.5 MiB [3][512][512]

  prep_weights<<<(3 * 512 * 512) / 256, 256, 0, stream>>>(Wq, Wk, Wv, Wt);
  qkv_proj<<<12288, 256, 0, stream>>>(x, Wt, bq, bk, bv, Qb, Kb, Vt);
  flash_attn<<<NBATCH * (SLEN / 16), 256, 0, stream>>>(Qb, Kb, Vt, out);
}
